// CAPTNet_5729486373541
// MI455X (gfx1250) — matrix operand order repaired and checked by reading, not compiled, not run on hardware
//
#include <hip/hip_runtime.h>

// ---------------------------------------------------------------------------
// CAPTNet block for MI455X (gfx1250, wave32, WMMA bf16 w/ f32 accumulate)
// Memory-bound workload (~0.7GB traffic @23.3TB/s ~ 30us); bf16 intermediates
// halve HBM traffic, WMMA 16x16x32_bf16 does all channel-mixing GEMMs.
// ---------------------------------------------------------------------------

typedef __attribute__((ext_vector_type(16))) __bf16 v16bf;
typedef __attribute__((ext_vector_type(8)))  __bf16 v8bf;
typedef __attribute__((ext_vector_type(8)))  float  v8f;

#define DEVI __device__ __forceinline__

constexpr int NPIX = 65536;   // 256*256
constexpr int BATCH = 8;

static DEVI float  tof(__bf16 v) { return (float)v; }
static DEVI __bf16 tobf(float v) { return (__bf16)v; }

static DEVI v16bf cat16(v8bf a, v8bf b) {
    v16bf r;
#pragma unroll
    for (int i = 0; i < 8; ++i) { r[i] = a[i]; r[i + 8] = b[i]; }
    return r;
}

// A operand (16x32 bf16, MxK): lane<16 holds K {0..7,16..23}, lane>=16 holds
// K {8..15,24..31}; row-major weight tile w[M][kpad], 16B-aligned runs.
static DEVI v16bf wmmaA(const __bf16* w, int kpad, int row, int ks, int lane) {
    int lo = (lane & 16) ? 8 : 0;
    const __bf16* p = w + row * kpad + ks * 32 + lo;
    return cat16(*(const v8bf*)p, *(const v8bf*)(p + 16));
}

// B operand (32x16 bf16, KxN): lane<16 holds K {0..7,16..23}, lane>=16 K
// {8..15,24..31} for column N=lane%16; tiles staged pixel-major t[px][kpad],
// 16B-aligned runs.
static DEVI v16bf wmmaB(const __bf16* t, int kpad, int px, int ks, int lane) {
    int ko = (lane & 16) ? 8 : 0;
    const __bf16* p = t + px * kpad + ks * 32 + ko;
    return cat16(*(const v8bf*)p, *(const v8bf*)(p + 16));
}

static DEVI v8f wmma_bf16(v16bf a, v16bf b, v8f c) {
    return __builtin_amdgcn_wmma_f32_16x16x32_bf16(false, a, false, b,
                                                   (short)0, c, false, false);
}

static DEVI void zero8(v8f& a) {
#pragma unroll
    for (int j = 0; j < 8; ++j) a[j] = 0.f;
}

// ---------------------------------------------------------------------------
// K1: xp = in_proj(x);  qkv_pre = qkv_w @ LN1(xp) + qkv_b   (WMMA, M=144,K=48)
// ---------------------------------------------------------------------------
__global__ __launch_bounds__(256) void k_in_ln_qkv(
    const float* __restrict__ x, const float* __restrict__ win,
    const float* __restrict__ bin, const float* __restrict__ lw,
    const float* __restrict__ lb, const float* __restrict__ qw,
    const float* __restrict__ qb, float* __restrict__ xp,
    __bf16* __restrict__ qkv_pre)
{
    __shared__ __align__(16) __bf16 lnT[128 * 64];   // [px][K] LN'ed, K-padded
    __shared__ __align__(16) __bf16 wT[144 * 64];    // qkv weights, K-padded
    const int tid = threadIdx.x;
    const int b = blockIdx.y;
    const int n0 = blockIdx.x * 128;

    for (int i = tid; i < 144 * 64; i += 256) {
        int r = i >> 6, k = i & 63;
        wT[i] = tobf((k < 48) ? qw[r * 48 + k] : 0.f);
    }
    if (tid < 128) {
        const int n = n0 + tid;
        const float* xb = x + (size_t)b * 3 * NPIX + n;
        float x0 = xb[0], x1 = xb[NPIX], x2 = xb[2 * NPIX];
        float v[48];
        float mu = 0.f;
#pragma unroll
        for (int c = 0; c < 48; ++c) {
            v[c] = win[c * 3] * x0 + win[c * 3 + 1] * x1 + win[c * 3 + 2] * x2 + bin[c];
            mu += v[c];
        }
        mu *= (1.f / 48.f);
        float var = 0.f;
#pragma unroll
        for (int c = 0; c < 48; ++c) { float d = v[c] - mu; var += d * d; }
        var *= (1.f / 48.f);
        float rstd = rsqrtf(var + 1e-6f);
#pragma unroll
        for (int c = 0; c < 48; ++c) {
            xp[((size_t)b * 48 + c) * NPIX + n] = v[c];
            lnT[tid * 64 + c] = tobf(lw[c] * (v[c] - mu) * rstd + lb[c]);
        }
#pragma unroll
        for (int c = 48; c < 64; ++c) lnT[tid * 64 + c] = tobf(0.f);
    }
    __syncthreads();

    const int lane = tid & 31, wv = tid >> 5;
    v8f acc[9];
#pragma unroll
    for (int m = 0; m < 9; ++m) zero8(acc[m]);
#pragma unroll
    for (int ks = 0; ks < 2; ++ks) {
        v16bf Bv = wmmaB(lnT, 64, wv * 16 + (lane & 15), ks, lane);
#pragma unroll
        for (int m = 0; m < 9; ++m) {
            v16bf Av = wmmaA(wT, 64, m * 16 + (lane & 15), ks, lane);
            acc[m] = wmma_bf16(Av, Bv, acc[m]);
        }
    }
    const int n = n0 + wv * 16 + (lane & 15);
    const int chi = (lane & 16) ? 8 : 0;
#pragma unroll
    for (int m = 0; m < 9; ++m)
#pragma unroll
        for (int j = 0; j < 8; ++j) {
            int c = m * 16 + j + chi;
            qkv_pre[((size_t)b * 144 + c) * NPIX + n] = tobf(acc[m][j] + qb[c]);
        }
}

// ---------------------------------------------------------------------------
// K2: depthwise 3x3 (SAME, zero pad) over qkv_pre -> qkv_dw   (144 channels)
// ---------------------------------------------------------------------------
__global__ __launch_bounds__(256) void k_dw_qkv(
    const __bf16* __restrict__ src, const float* __restrict__ w,
    const float* __restrict__ bias, __bf16* __restrict__ dst)
{
    int idx = blockIdx.x * 256 + threadIdx.x;  // [b*144+c][n] flat
    int n = idx & (NPIX - 1);
    int bc = idx >> 16;
    int c = bc % 144;
    int y = n >> 8, xc = n & 255;
    float s = bias[c];
    const __bf16* p = src + ((size_t)bc << 16);
#pragma unroll
    for (int ky = -1; ky <= 1; ++ky) {
        int yy = y + ky;
        if (yy < 0 || yy > 255) continue;
#pragma unroll
        for (int kx = -1; kx <= 1; ++kx) {
            int xx = xc + kx;
            if (xx < 0 || xx > 255) continue;
            s += w[c * 9 + (ky + 1) * 3 + (kx + 1)] * tof(p[yy * 256 + xx]);
        }
    }
    dst[idx] = tobf(s);
}

// ---------------------------------------------------------------------------
// K3: per-(b,h,chunk) partial Gram G=q k^T plus |q|^2,|k|^2 (deterministic)
// ---------------------------------------------------------------------------
__global__ __launch_bounds__(256) void k_gram(
    const __bf16* __restrict__ qkv, float* __restrict__ partial)
{
    __shared__ __bf16 qs[12 * 64];
    __shared__ __bf16 ksh[12 * 64];
    const int tid = threadIdx.x;
    const int bh = blockIdx.x >> 6, chunk = blockIdx.x & 63;
    const int b = bh >> 2, h = bh & 3;
    const __bf16* qp = qkv + ((size_t)b * 144 + h * 12) * NPIX;
    const __bf16* kp = qkv + ((size_t)b * 144 + 48 + h * 12) * NPIX;
    float acc = 0.f;
    for (int it = 0; it < 16; ++it) {
        int off = chunk * 1024 + it * 64;
        for (int i = tid; i < 768; i += 256) {
            int cc = i >> 6, pp = i & 63;
            qs[i] = qp[(size_t)cc * NPIX + off + pp];
            ksh[i] = kp[(size_t)cc * NPIX + off + pp];
        }
        __syncthreads();
        if (tid < 144) {
            int c = tid / 12, d = tid % 12;
            for (int p = 0; p < 64; ++p)
                acc += tof(qs[c * 64 + p]) * tof(ksh[d * 64 + p]);
        } else if (tid < 156) {
            int c = tid - 144;
            for (int p = 0; p < 64; ++p) { float q = tof(qs[c * 64 + p]); acc += q * q; }
        } else if (tid < 168) {
            int c = tid - 156;
            for (int p = 0; p < 64; ++p) { float k = tof(ksh[c * 64 + p]); acc += k * k; }
        }
        __syncthreads();
    }
    if (tid < 168) partial[((size_t)bh * 64 + chunk) * 168 + tid] = acc;
}

// ---------------------------------------------------------------------------
// K4: reduce partials (fixed order), softmax, Wc[b] = attn_out_w @ blockdiag(attn)
// ---------------------------------------------------------------------------
__global__ __launch_bounds__(256) void k_attn_wc(
    const float* __restrict__ partial, const float* __restrict__ temp,
    const float* __restrict__ wout, __bf16* __restrict__ Wc)
{
    __shared__ float G[4][12][12];
    __shared__ float qn[4][12], kn[4][12];
    __shared__ float attn[4][12][12];
    const int tid = threadIdx.x;
    const int b = blockIdx.x;
    for (int e = tid; e < 672; e += 256) {
        int h = e / 168, idx = e % 168;
        const float* p = partial + ((size_t)(b * 4 + h) * 64) * 168 + idx;
        float s = 0.f;
        for (int ch = 0; ch < 64; ++ch) s += p[ch * 168];
        if (idx < 144) G[h][idx / 12][idx % 12] = s;
        else if (idx < 156) qn[h][idx - 144] = fmaxf(sqrtf(s), 1e-12f);
        else kn[h][idx - 156] = fmaxf(sqrtf(s), 1e-12f);
    }
    __syncthreads();
    if (tid < 48) {
        int h = tid / 12, c = tid % 12;
        float t = temp[h];
        float a[12], mx = -1e30f;
        for (int d = 0; d < 12; ++d) {
            a[d] = G[h][c][d] / (qn[h][c] * kn[h][d]) * t;
            mx = fmaxf(mx, a[d]);
        }
        float sum = 0.f;
        for (int d = 0; d < 12; ++d) { a[d] = expf(a[d] - mx); sum += a[d]; }
        float inv = 1.f / sum;
        for (int d = 0; d < 12; ++d) attn[h][c][d] = a[d] * inv;
    }
    __syncthreads();
    for (int e = tid; e < 48 * 64; e += 256) {   // K padded 48 -> 64
        int o = e >> 6, dg = e & 63;
        float s = 0.f;
        if (dg < 48) {
            int h = dg / 12, d = dg % 12;
            for (int c = 0; c < 12; ++c)
                s += wout[o * 48 + h * 12 + c] * attn[h][c][d];
        }
        Wc[(size_t)b * 48 * 64 + e] = tobf(s);
    }
}

// ---------------------------------------------------------------------------
// K5: xp += Wc[b] @ v + attn_out_b    (WMMA, M=48, K=48->64)
// ---------------------------------------------------------------------------
__global__ __launch_bounds__(256) void k_attnv(
    const __bf16* __restrict__ Wc, const __bf16* __restrict__ qkv,
    const float* __restrict__ bias, float* __restrict__ xp)
{
    __shared__ __align__(16) __bf16 wcT[48 * 64];
    __shared__ __align__(16) __bf16 vT[128 * 64];
    const int tid = threadIdx.x;
    const int b = blockIdx.y;
    const int n0 = blockIdx.x * 128;
    for (int i = tid; i < 48 * 64; i += 256) wcT[i] = Wc[(size_t)b * 48 * 64 + i];
    for (int i = tid; i < 128 * 64; i += 256) {
        int k = i >> 7, px = i & 127;
        __bf16 v = tobf(0.f);
        if (k < 48) v = qkv[((size_t)b * 144 + 96 + k) * NPIX + n0 + px];
        vT[px * 64 + k] = v;
    }
    __syncthreads();
    const int lane = tid & 31, wv = tid >> 5;
    v8f acc[3];
#pragma unroll
    for (int m = 0; m < 3; ++m) zero8(acc[m]);
#pragma unroll
    for (int ks = 0; ks < 2; ++ks) {
        v16bf Bv = wmmaB(vT, 64, wv * 16 + (lane & 15), ks, lane);
#pragma unroll
        for (int m = 0; m < 3; ++m) {
            v16bf Av = wmmaA(wcT, 64, m * 16 + (lane & 15), ks, lane);
            acc[m] = wmma_bf16(Av, Bv, acc[m]);
        }
    }
    const int n = n0 + wv * 16 + (lane & 15);
    const int chi = (lane & 16) ? 8 : 0;
#pragma unroll
    for (int m = 0; m < 3; ++m)
#pragma unroll
        for (int j = 0; j < 8; ++j) {
            int c = m * 16 + j + chi;
            size_t gi = ((size_t)b * 48 + c) * NPIX + n;
            xp[gi] += acc[m][j] + bias[c];
        }
}

// ---------------------------------------------------------------------------
// K6: ffn_pre = ffn_in_w @ LN2(xp) + ffn_in_b    (WMMA, M=192, K=48->64)
// ---------------------------------------------------------------------------
__global__ __launch_bounds__(256) void k_ln_ffn(
    const float* __restrict__ xp, const float* __restrict__ lw,
    const float* __restrict__ lb, const float* __restrict__ fw,
    const float* __restrict__ fb, __bf16* __restrict__ ffn_pre)
{
    __shared__ __align__(16) __bf16 lnT[128 * 64];
    __shared__ __align__(16) __bf16 wT[192 * 64];
    const int tid = threadIdx.x;
    const int b = blockIdx.y;
    const int n0 = blockIdx.x * 128;
    for (int i = tid; i < 192 * 64; i += 256) {
        int r = i >> 6, k = i & 63;
        wT[i] = tobf((k < 48) ? fw[r * 48 + k] : 0.f);
    }
    if (tid < 128) {
        const int n = n0 + tid;
        float v[48];
        float mu = 0.f;
#pragma unroll
        for (int c = 0; c < 48; ++c) {
            v[c] = xp[((size_t)b * 48 + c) * NPIX + n];
            mu += v[c];
        }
        mu *= (1.f / 48.f);
        float var = 0.f;
#pragma unroll
        for (int c = 0; c < 48; ++c) { float d = v[c] - mu; var += d * d; }
        var *= (1.f / 48.f);
        float rstd = rsqrtf(var + 1e-6f);
#pragma unroll
        for (int c = 0; c < 48; ++c)
            lnT[tid * 64 + c] = tobf(lw[c] * (v[c] - mu) * rstd + lb[c]);
#pragma unroll
        for (int c = 48; c < 64; ++c) lnT[tid * 64 + c] = tobf(0.f);
    }
    __syncthreads();
    const int lane = tid & 31, wv = tid >> 5;
    v8f acc[12];
#pragma unroll
    for (int m = 0; m < 12; ++m) zero8(acc[m]);
#pragma unroll
    for (int ks = 0; ks < 2; ++ks) {
        v16bf Bv = wmmaB(lnT, 64, wv * 16 + (lane & 15), ks, lane);
#pragma unroll
        for (int m = 0; m < 12; ++m) {
            v16bf Av = wmmaA(wT, 64, m * 16 + (lane & 15), ks, lane);
            acc[m] = wmma_bf16(Av, Bv, acc[m]);
        }
    }
    const int n = n0 + wv * 16 + (lane & 15);
    const int chi = (lane & 16) ? 8 : 0;
#pragma unroll
    for (int m = 0; m < 12; ++m)
#pragma unroll
        for (int j = 0; j < 8; ++j) {
            int c = m * 16 + j + chi;
            ffn_pre[((size_t)b * 192 + c) * NPIX + n] = tobf(acc[m][j] + fb[c]);
        }
}

// ---------------------------------------------------------------------------
// K7: depthwise 3x3 over ffn_pre (192ch) + SimpleGate -> gate (96ch)
// ---------------------------------------------------------------------------
__global__ __launch_bounds__(256) void k_dw_gate(
    const __bf16* __restrict__ src, const float* __restrict__ w,
    const float* __restrict__ bias, __bf16* __restrict__ gate)
{
    int idx = blockIdx.x * 256 + threadIdx.x;  // [b*96+c][n] flat
    int n = idx & (NPIX - 1);
    int bc = idx >> 16;
    int c = bc % 96, b = bc / 96;
    int y = n >> 8, xc = n & 255;
    float s1 = bias[c], s2 = bias[c + 96];
    const __bf16* p1 = src + ((size_t)b * 192 + c) * NPIX;
    const __bf16* p2 = src + ((size_t)b * 192 + 96 + c) * NPIX;
#pragma unroll
    for (int ky = -1; ky <= 1; ++ky) {
        int yy = y + ky;
        if (yy < 0 || yy > 255) continue;
#pragma unroll
        for (int kx = -1; kx <= 1; ++kx) {
            int xx = xc + kx;
            if (xx < 0 || xx > 255) continue;
            int wi = (ky + 1) * 3 + (kx + 1);
            s1 += w[c * 9 + wi] * tof(p1[yy * 256 + xx]);
            s2 += w[(c + 96) * 9 + wi] * tof(p2[yy * 256 + xx]);
        }
    }
    gate[idx] = tobf(s1 * s2);
}

// ---------------------------------------------------------------------------
// K8: y = ffn_out_w @ gate + ffn_out_b; xpf = xp + y; out = out_proj(xpf)
//     (WMMA, M=48, K=96 exact; out_proj 3x48 in VALU via LDS)
// ---------------------------------------------------------------------------
__global__ __launch_bounds__(256) void k_ffn_out(
    const __bf16* __restrict__ gate, const float* __restrict__ wf,
    const float* __restrict__ bf, const float* __restrict__ xp,
    const float* __restrict__ wo, const float* __restrict__ bo,
    float* __restrict__ out)
{
    __shared__ __align__(16) __bf16 wT[48 * 96];
    __shared__ __align__(16) __bf16 gT[128 * 96];
    __shared__ float xpf[128 * 48];
    const int tid = threadIdx.x;
    const int b = blockIdx.y;
    const int n0 = blockIdx.x * 128;
    for (int i = tid; i < 48 * 96; i += 256) wT[i] = tobf(wf[i]);
    for (int i = tid; i < 128 * 96; i += 256) {
        int k = i >> 7, px = i & 127;
        gT[px * 96 + k] = gate[((size_t)b * 96 + k) * NPIX + n0 + px];
    }
    __syncthreads();
    const int lane = tid & 31, wv = tid >> 5;
    v8f acc[3];
#pragma unroll
    for (int m = 0; m < 3; ++m) zero8(acc[m]);
#pragma unroll
    for (int ks = 0; ks < 3; ++ks) {
        v16bf Bv = wmmaB(gT, 96, wv * 16 + (lane & 15), ks, lane);
#pragma unroll
        for (int m = 0; m < 3; ++m) {
            v16bf Av = wmmaA(wT, 96, m * 16 + (lane & 15), ks, lane);
            acc[m] = wmma_bf16(Av, Bv, acc[m]);
        }
    }
    const int pxl = wv * 16 + (lane & 15);
    const int n = n0 + pxl;
    const int chi = (lane & 16) ? 8 : 0;
#pragma unroll
    for (int m = 0; m < 3; ++m)
#pragma unroll
        for (int j = 0; j < 8; ++j) {
            int c = m * 16 + j + chi;
            float v = acc[m][j] + bf[c] + xp[((size_t)b * 48 + c) * NPIX + n];
            xpf[pxl * 48 + c] = v;
        }
    __syncthreads();
    if (tid < 128) {
        int n2 = n0 + tid;
        for (int o = 0; o < 3; ++o) {
            float s = bo[o];
            for (int c = 0; c < 48; ++c) s += wo[o * 48 + c] * xpf[tid * 48 + c];
            out[((size_t)b * 3 + o) * NPIX + n2] = s;
        }
    }
}

// ---------------------------------------------------------------------------
// Workspace layout (bytes). Total ~404 MB; ffn_pre/gate alias dead qkv bufs.
// ---------------------------------------------------------------------------
constexpr size_t SZ_XP      = (size_t)BATCH * 48 * NPIX * 4;   // 100,663,296
constexpr size_t SZ_QKV     = (size_t)BATCH * 144 * NPIX * 2;  // 150,994,944
constexpr size_t OFF_XP     = 0;
constexpr size_t OFF_QKVP   = OFF_XP + SZ_XP;
constexpr size_t OFF_QKVD   = OFF_QKVP + SZ_QKV;
constexpr size_t OFF_PART   = OFF_QKVD + SZ_QKV;
constexpr size_t SZ_PART    = (size_t)32 * 64 * 168 * 4;
constexpr size_t OFF_WC     = OFF_PART + SZ_PART;
constexpr size_t SZ_FFN_PRE = (size_t)BATCH * 192 * NPIX * 2;  // 201,326,592
constexpr size_t OFF_FFNP   = OFF_QKVP;                // alias (qkv dead)
constexpr size_t OFF_GATE   = OFF_QKVP + SZ_FFN_PRE;   // fits before OFF_PART

extern "C" void kernel_launch(void* const* d_in, const int* in_sizes, int n_in,
                              void* d_out, int out_size, void* d_ws, size_t ws_size,
                              hipStream_t stream)
{
    const float* x          = (const float*)d_in[0];
    const float* in_proj_w  = (const float*)d_in[1];
    const float* in_proj_b  = (const float*)d_in[2];
    const float* ln1_w      = (const float*)d_in[3];
    const float* ln1_b      = (const float*)d_in[4];
    const float* qkv_w      = (const float*)d_in[5];
    const float* qkv_b      = (const float*)d_in[6];
    const float* qkv_dw_w   = (const float*)d_in[7];
    const float* qkv_dw_b   = (const float*)d_in[8];
    const float* temperature= (const float*)d_in[9];
    const float* attn_out_w = (const float*)d_in[10];
    const float* attn_out_b = (const float*)d_in[11];
    const float* ln2_w      = (const float*)d_in[12];
    const float* ln2_b      = (const float*)d_in[13];
    const float* ffn_in_w   = (const float*)d_in[14];
    const float* ffn_in_b   = (const float*)d_in[15];
    const float* ffn_dw_w   = (const float*)d_in[16];
    const float* ffn_dw_b   = (const float*)d_in[17];
    const float* ffn_out_w  = (const float*)d_in[18];
    const float* ffn_out_b  = (const float*)d_in[19];
    const float* out_proj_w = (const float*)d_in[20];
    const float* out_proj_b = (const float*)d_in[21];

    char* ws = (char*)d_ws;
    float*  xp       = (float*)(ws + OFF_XP);
    __bf16* qkv_pre  = (__bf16*)(ws + OFF_QKVP);
    __bf16* qkv_dw   = (__bf16*)(ws + OFF_QKVD);
    float*  partial  = (float*)(ws + OFF_PART);
    __bf16* Wc       = (__bf16*)(ws + OFF_WC);
    __bf16* ffn_pre  = (__bf16*)(ws + OFF_FFNP);
    __bf16* gate     = (__bf16*)(ws + OFF_GATE);
    float*  out      = (float*)d_out;

    dim3 gTile(512, BATCH);   // 512 tiles * 128 px = 65536 pixels per batch

    k_in_ln_qkv<<<gTile, 256, 0, stream>>>(x, in_proj_w, in_proj_b, ln1_w, ln1_b,
                                           qkv_w, qkv_b, xp, qkv_pre);
    k_dw_qkv<<<(BATCH * 144 * NPIX) / 256, 256, 0, stream>>>(qkv_pre, qkv_dw_w,
                                                             qkv_dw_b, qkv_dw);
    k_gram<<<32 * 64, 256, 0, stream>>>(qkv_dw, partial);
    k_attn_wc<<<BATCH, 256, 0, stream>>>(partial, temperature, attn_out_w, Wc);
    k_attnv<<<gTile, 256, 0, stream>>>(Wc, qkv_dw, attn_out_b, xp);
    k_ln_ffn<<<gTile, 256, 0, stream>>>(xp, ln2_w, ln2_b, ffn_in_w, ffn_in_b,
                                        ffn_pre);
    k_dw_gate<<<(BATCH * 96 * NPIX) / 256, 256, 0, stream>>>(ffn_pre, ffn_dw_w,
                                                             ffn_dw_b, gate);
    k_ffn_out<<<gTile, 256, 0, stream>>>(gate, ffn_out_w, ffn_out_b, xp,
                                         out_proj_w, out_proj_b, out);
}
